// SpatioTemporalGNN_11785390260851
// MI455X (gfx1250) — matrix operand order repaired and checked by reading, not compiled, not run on hardware
//
#include <hip/hip_runtime.h>
#include <hip/hip_bf16.h>
#include <stdint.h>

// ---------------------------------------------------------------------------
// SpatioTemporalGNN for MI455X (gfx1250): bf16 WMMA GEMM pipeline
//   B=64 T=32 M=64 IN=256 GNN=256 H=4 C=64 L=3 TEMP=256 OUT=512 NH=8 NLT=2
// ---------------------------------------------------------------------------

typedef unsigned short u16;
typedef __bf16  bf16x16 __attribute__((ext_vector_type(16)));
typedef float   f32x8   __attribute__((ext_vector_type(8)));

union Frag { bf16x16 v; uint4 q[2]; };

#define GFLAG_RELU    1
#define GFLAG_ADDC    2
#define GFLAG_MIR_T64 4

__device__ __forceinline__ u16 f2bf(float f) {
    uint32_t u = __float_as_uint(f);
    uint32_t r = u + 0x7fffu + ((u >> 16) & 1u);   // round-to-nearest-even
    return (u16)(r >> 16);
}

__device__ __forceinline__ float wave_sum(float v) {
    #pragma unroll
    for (int off = 16; off; off >>= 1) v += __shfl_xor(v, off, 32);
    return v;
}

// ---------------------------------------------------------------------------
// f32 -> bf16 conversion (plain, and transposed for square per-layer weights)
// ---------------------------------------------------------------------------
__global__ void cvt_bf16_kernel(const float* __restrict__ src, u16* __restrict__ dst, int n) {
    int i = blockIdx.x * blockDim.x + threadIdx.x;
    if (i < n) dst[i] = f2bf(src[i]);
}

// dst[l][k][g] = src[l][g][k] for L layers of DxD matrices (D=256 here).
__global__ void cvt_bf16_t_kernel(const float* __restrict__ src, u16* __restrict__ dst,
                                  int D, int n) {
    int i = blockIdx.x * blockDim.x + threadIdx.x;
    if (i >= n) return;
    int l = i / (D * D);
    int r = i - l * (D * D);
    int g = r / D, k = r - g * D;
    dst[(size_t)l * D * D + (size_t)k * D + g] = f2bf(src[i]);
}

// ---------------------------------------------------------------------------
// Generic WMMA GEMM:  C[R,N] (f32, row-major) = A[R,K](bf16) * W[N,K](bf16)^T
// One 32x64 output tile per wave (2 A-frags x 4 B-frags -> 8 wmma per K-step).
// Requires: K % 32 == 0, R % 32 == 0, N % 64 == 0.
// Epilogue: +bias[col], +=C (residual), *rowScale(row, binarized), relu,
//           optional bf16 mirror (row-major or per-64-row-group transposed).
// ---------------------------------------------------------------------------
__global__ __launch_bounds__(256) void gemm_bf16_nt(
    const u16* __restrict__ A, const u16* __restrict__ W,
    const float* __restrict__ bias, const float* __restrict__ rowScale,
    float* __restrict__ C, u16* __restrict__ Cbf,
    int R, int N, int K, int flags)
{
    int wid  = threadIdx.x >> 5;
    int lane = threadIdx.x & 31;
    int nct  = N >> 6;                 // 64-wide column tiles
    int tiles = (R >> 5) * nct;        // 32-tall row tiles
    int tile = blockIdx.x * (blockDim.x >> 5) + wid;
    if (tile >= tiles) return;
    int rt = tile / nct, ct = tile % nct;

    int  sub = lane & 15;
    bool hi  = (lane & 16) != 0;

    const size_t rstep = (size_t)16 * K;
    const u16* Ap = A + (size_t)((rt << 5) + sub) * K + (hi ? 8 : 0);
    const u16* Wp = W + (size_t)((ct << 6) + sub) * K + (hi ? 8 : 0);

    f32x8 acc[8];
    #pragma unroll
    for (int i = 0; i < 8; i++)
        #pragma unroll
        for (int e = 0; e < 8; e++) acc[i][e] = 0.0f;

    for (int k0 = 0; k0 < K; k0 += 32) {
        Frag a0, a1;
        a0.q[0] = *(const uint4*)(Ap + k0);
        a0.q[1] = *(const uint4*)(Ap + k0 + 16);
        a1.q[0] = *(const uint4*)(Ap + rstep + k0);
        a1.q[1] = *(const uint4*)(Ap + rstep + k0 + 16);
        __builtin_prefetch(Ap + k0 + 32, 0, 3);           // global_prefetch_b8
        __builtin_prefetch(Ap + rstep + k0 + 32, 0, 3);
        #pragma unroll
        for (int j = 0; j < 4; j++) {
            const u16* bp = Wp + (size_t)j * rstep + k0;
            Frag b;
            b.q[0] = *(const uint4*)(bp);
            b.q[1] = *(const uint4*)(bp + 16);
            acc[j]     = __builtin_amdgcn_wmma_f32_16x16x32_bf16(
                             false, a0.v, false, b.v, (short)0, acc[j],     false, false);
            acc[4 + j] = __builtin_amdgcn_wmma_f32_16x16x32_bf16(
                             false, a1.v, false, b.v, (short)0, acc[4 + j], false, false);
        }
    }

    #pragma unroll
    for (int j = 0; j < 4; j++) {
        int col = (ct << 6) + j * 16 + sub;
        float bv = bias ? bias[col] : 0.0f;
        #pragma unroll
        for (int i = 0; i < 2; i++) {
            int rbase = (rt << 5) + i * 16 + (hi ? 8 : 0);
            #pragma unroll
            for (int r = 0; r < 8; r++) {
                int row = rbase + r;
                size_t idx = (size_t)row * N + col;
                float v = acc[i * 4 + j][r] + bv;
                if (flags & GFLAG_ADDC) v += C[idx];
                if (rowScale) v *= (rowScale[row] > 0.5f ? 1.0f : 0.0f);
                if (flags & GFLAG_RELU) v = fmaxf(v, 0.0f);
                C[idx] = v;
                if (Cbf) {
                    if (flags & GFLAG_MIR_T64) {
                        int bt = row >> 6, m = row & 63;
                        Cbf[((size_t)bt * N + col) * 64 + m] = f2bf(v);
                    } else {
                        Cbf[idx] = f2bf(v);
                    }
                }
            }
        }
    }
}

// ---------------------------------------------------------------------------
// GAT attention coefficients. One block per (b,t); thread = (dest d, head h).
// Writes unnormalized exp(logit-max) as bf16 in [bt][h][d][s] layout + 1/sum.
// ---------------------------------------------------------------------------
__global__ __launch_bounds__(256) void gat_alpha(
    const float* __restrict__ hF, const float* __restrict__ boxes,
    const float* __restrict__ mask,
    const float* __restrict__ attSrc, const float* __restrict__ attDst,
    const float* __restrict__ attEdge, const float* __restrict__ edgeW,
    u16* __restrict__ alphaT, float* __restrict__ invsum)
{
    int bt = blockIdx.x, tid = threadIdx.x;
    __shared__ float px[64], py[64], vld[64];
    __shared__ float asrc[64][4], adst[64][4];
    __shared__ float lattr[64][3];
    __shared__ float we[3][4];

    if (tid < 64) {
        const float* bx = boxes + ((size_t)bt * 64 + tid) * 5;
        px[tid] = bx[1]; py[tid] = bx[2];
        vld[tid] = (mask[(size_t)bt * 64 + tid] > 0.5f) ? 1.0f : 0.0f;
    }
    if (tid >= 64 && tid < 76) {
        int e = (tid - 64) >> 2, h = (tid - 64) & 3;
        float s = 0.0f;
        for (int c = 0; c < 64; c++) s += edgeW[e * 256 + h * 64 + c] * attEdge[h * 64 + c];
        we[e][h] = s;
    }
    __syncthreads();

    {   // a_src / a_dst for (m,h)
        int m = tid >> 2, h = tid & 3;
        const float* hp = hF + ((size_t)bt * 64 + m) * 256 + h * 64;
        float s = 0.0f, dd = 0.0f;
        for (int c = 0; c < 64; c++) { float x = hp[c]; s += x * attSrc[h * 64 + c]; dd += x * attDst[h * 64 + c]; }
        asrc[m][h] = s; adst[m][h] = dd;
    }
    if (tid < 64) {   // self-loop attributes for dest d
        int d = tid; float sd = 0, sx = 0, sy = 0; int cnt = 0;
        for (int s = 0; s < 64; s++) {
            if (s == d) continue;
            float rx = px[d] - px[s], ry = py[d] - py[s];
            float dist = sqrtf(fmaxf(rx * rx + ry * ry, 1e-12f));
            if (vld[s] > 0.0f && vld[d] > 0.0f && dist < 0.3f) { sd += dist; sx += rx; sy += ry; cnt++; }
        }
        float dg = (float)(cnt > 1 ? cnt : 1);
        lattr[d][0] = sd / dg; lattr[d][1] = sx / dg; lattr[d][2] = sy / dg;
    }
    __syncthreads();

    int d = tid >> 2, h = tid & 3;
    float adh = adst[d][h];
    float w0 = we[0][h], w1 = we[1][h], w2 = we[2][h];

    auto logit = [&](int s) -> float {
        float ea0, ea1, ea2, conn;
        if (s == d) { ea0 = lattr[d][0]; ea1 = lattr[d][1]; ea2 = lattr[d][2]; conn = vld[d]; }
        else {
            float rx = px[d] - px[s], ry = py[d] - py[s];
            float dist = sqrtf(fmaxf(rx * rx + ry * ry, 1e-12f));
            float a = (vld[s] > 0.0f && vld[d] > 0.0f && dist < 0.3f) ? 1.0f : 0.0f;
            ea0 = a * dist; ea1 = a * rx; ea2 = a * ry; conn = a;
        }
        float lg = asrc[s][h] + adh + ea0 * w0 + ea1 * w1 + ea2 * w2;
        lg = (lg >= 0.0f) ? lg : 0.2f * lg;
        return (conn > 0.0f) ? lg : -1e9f;
    };

    float mx = -3.4e38f;
    for (int s = 0; s < 64; s++) mx = fmaxf(mx, logit(s));
    u16* ap = alphaT + ((size_t)(bt * 4 + h) * 64 + d) * 64;
    float sum = 0.0f;
    for (int s = 0; s < 64; s++) {
        float e = __expf(logit(s) - mx);
        sum += e;
        ap[s] = f2bf(e);
    }
    invsum[(size_t)(bt * 4 + h) * 64 + d] = 1.0f / sum;
}

// ---------------------------------------------------------------------------
// GAT message aggregation via WMMA: per (b,t,head) one wave computes
//   msg[d, h*64+c] = (1/sum[d,h]) * sum_s alpha[s,d,h] * h[s, h*64+c]
// A = alphaT [bt][h][d][s] bf16 (rows contiguous in s);
// B = hT     [bt][col][m]  bf16 (rows contiguous in s).
// All 8 A-fragments cached in registers -> 32 b128 loads for 32 wmma.
// ---------------------------------------------------------------------------
__global__ __launch_bounds__(256) void gat_msg(
    const u16* __restrict__ alphaT, const u16* __restrict__ hT,
    const float* __restrict__ invsum, float* __restrict__ msg)
{
    int wid  = threadIdx.x >> 5;
    int lane = threadIdx.x & 31;
    int w = blockIdx.x * (blockDim.x >> 5) + wid;
    int bt = w >> 2, head = w & 3;
    int  sub = lane & 15;
    bool hi  = (lane & 16) != 0;

    const u16* Abase = alphaT + (size_t)(bt * 4 + head) * 4096;
    const u16* Hbase = hT + ((size_t)bt * 256 + head * 64) * 64;
    const float* isp = invsum + (size_t)(bt * 4 + head) * 64;

    Frag a[4][2];   // [dTile][k-half]
    #pragma unroll
    for (int dt = 0; dt < 4; dt++) {
        const u16* Ap = Abase + (size_t)(dt * 16 + sub) * 64 + (hi ? 8 : 0);
        a[dt][0].q[0] = *(const uint4*)(Ap);
        a[dt][0].q[1] = *(const uint4*)(Ap + 16);
        a[dt][1].q[0] = *(const uint4*)(Ap + 32);
        a[dt][1].q[1] = *(const uint4*)(Ap + 48);
    }

    #pragma unroll
    for (int ctl = 0; ctl < 4; ctl++) {
        const u16* Hp = Hbase + (size_t)(ctl * 16 + sub) * 64 + (hi ? 8 : 0);
        Frag b0, b1;
        b0.q[0] = *(const uint4*)(Hp);
        b0.q[1] = *(const uint4*)(Hp + 16);
        b1.q[0] = *(const uint4*)(Hp + 32);
        b1.q[1] = *(const uint4*)(Hp + 48);
        int col = head * 64 + ctl * 16 + sub;
        #pragma unroll
        for (int dt = 0; dt < 4; dt++) {
            f32x8 acc;
            #pragma unroll
            for (int i = 0; i < 8; i++) acc[i] = 0.0f;
            acc = __builtin_amdgcn_wmma_f32_16x16x32_bf16(
                      false, a[dt][0].v, false, b0.v, (short)0, acc, false, false);
            acc = __builtin_amdgcn_wmma_f32_16x16x32_bf16(
                      false, a[dt][1].v, false, b1.v, (short)0, acc, false, false);
            int mb = dt * 16 + (hi ? 8 : 0);
            #pragma unroll
            for (int r = 0; r < 8; r++) {
                int m = mb + r;
                msg[((size_t)bt * 64 + m) * 256 + col] = acc[r] * isp[m];
            }
        }
    }
}

// ---------------------------------------------------------------------------
// GAT epilogue: x = relu(LN(msg + gat_bias + x)); wave per row (256 cols).
// ---------------------------------------------------------------------------
__global__ __launch_bounds__(256) void gat_epilogue(
    const float* __restrict__ msg, const float* __restrict__ bias,
    const float* __restrict__ lnS, const float* __restrict__ lnB,
    float* __restrict__ xF, u16* __restrict__ xBf, int rows)
{
    int wave = blockIdx.x * (blockDim.x >> 5) + (threadIdx.x >> 5);
    int lane = threadIdx.x & 31;
    if (wave >= rows) return;
    const float* mp = msg + (size_t)wave * 256;
    float* xp = xF + (size_t)wave * 256;
    float v[8]; float sum = 0.0f;
    #pragma unroll
    for (int j = 0; j < 8; j++) { int c = j * 32 + lane; v[j] = mp[c] + bias[c] + xp[c]; sum += v[j]; }
    float mean = wave_sum(sum) * (1.0f / 256.0f);
    float vs = 0.0f;
    #pragma unroll
    for (int j = 0; j < 8; j++) { float d = v[j] - mean; vs += d * d; }
    float inv = rsqrtf(wave_sum(vs) * (1.0f / 256.0f) + 1e-5f);
    #pragma unroll
    for (int j = 0; j < 8; j++) {
        int c = j * 32 + lane;
        float o = (v[j] - mean) * inv * lnS[c] + lnB[c];
        o = fmaxf(o, 0.0f);
        xp[c] = o;
        xBf[(size_t)wave * 256 + c] = f2bf(o);
    }
}

// ---------------------------------------------------------------------------
// Generic LayerNorm: wave per row; width in {256,512}; optional relu;
// optional f32 and/or bf16 outputs.
// ---------------------------------------------------------------------------
__global__ __launch_bounds__(256) void ln_rows(
    const float* __restrict__ in, const float* __restrict__ s, const float* __restrict__ b,
    float* __restrict__ outF, u16* __restrict__ outBf, int rows, int width, int relu)
{
    int wave = blockIdx.x * (blockDim.x >> 5) + (threadIdx.x >> 5);
    int lane = threadIdx.x & 31;
    if (wave >= rows) return;
    const float* rp = in + (size_t)wave * width;
    int n = width >> 5;
    float v[16]; float sum = 0.0f;
    for (int j = 0; j < n; j++) { v[j] = rp[j * 32 + lane]; sum += v[j]; }
    float mean = wave_sum(sum) / (float)width;
    float vs = 0.0f;
    for (int j = 0; j < n; j++) { float d = v[j] - mean; vs += d * d; }
    float inv = rsqrtf(wave_sum(vs) / (float)width + 1e-5f);
    for (int j = 0; j < n; j++) {
        int c = j * 32 + lane;
        float o = (v[j] - mean) * inv * s[c] + b[c];
        if (relu) o = fmaxf(o, 0.0f);
        if (outF)  outF[(size_t)wave * width + c] = o;
        if (outBf) outBf[(size_t)wave * width + c] = f2bf(o);
    }
}

// ---------------------------------------------------------------------------
// Masked mean over drones: frame[bt][c] = sum_m x*valid / max(nvalid,1)
// ---------------------------------------------------------------------------
__global__ __launch_bounds__(256) void frame_pool(
    const float* __restrict__ x, const float* __restrict__ mask, u16* __restrict__ frameBf)
{
    int bt = blockIdx.x, c = threadIdx.x;
    __shared__ float nv;
    if (c == 0) {
        float s = 0.0f;
        for (int m = 0; m < 64; m++) s += (mask[(size_t)bt * 64 + m] > 0.5f) ? 1.0f : 0.0f;
        nv = fmaxf(s, 1.0f);
    }
    __syncthreads();
    float acc = 0.0f;
    for (int m = 0; m < 64; m++) {
        float mk = (mask[(size_t)bt * 64 + m] > 0.5f) ? 1.0f : 0.0f;
        acc += x[((size_t)bt * 64 + m) * 256 + c] * mk;
    }
    frameBf[(size_t)bt * 256 + c] = f2bf(acc / nv);
}

__global__ void add_pos_kernel(float* __restrict__ y, const float* __restrict__ pos) {
    int i = blockIdx.x * blockDim.x + threadIdx.x;
    if (i < 2048 * 256) {
        int t = (i >> 8) & 31;
        y[i] += pos[t * 256 + (i & 255)];
    }
}

// ---------------------------------------------------------------------------
// Temporal self-attention: T=32, NH=8, hd=32. Block per batch b; thread=(h,tq).
// qkv layout [b*T+t][768] = [q(256) | k(256) | v(256)]. Writes o as bf16.
// ---------------------------------------------------------------------------
__global__ __launch_bounds__(256) void temporal_attn(
    const float* __restrict__ qkv, u16* __restrict__ oBf)
{
    int b = blockIdx.x;
    int h = threadIdx.x >> 5, tq = threadIdx.x & 31;
    const float scale = 0.17677669529663687f;   // 1/sqrt(32)
    float q[32], sc[32];
    const float* qp = qkv + (size_t)(b * 32 + tq) * 768 + h * 32;
    #pragma unroll
    for (int c = 0; c < 32; c++) q[c] = qp[c];
    float mx = -3.4e38f;
    for (int kt = 0; kt < 32; kt++) {
        const float* kp = qkv + (size_t)(b * 32 + kt) * 768 + 256 + h * 32;
        float d = 0.0f;
        #pragma unroll
        for (int c = 0; c < 32; c++) d += q[c] * kp[c];
        d *= scale;
        sc[kt] = d;
        mx = fmaxf(mx, d);
    }
    float se = 0.0f;
    #pragma unroll
    for (int kt = 0; kt < 32; kt++) { sc[kt] = __expf(sc[kt] - mx); se += sc[kt]; }
    float inv = 1.0f / se;
    for (int c = 0; c < 32; c++) {
        float o = 0.0f;
        for (int kt = 0; kt < 32; kt++)
            o += sc[kt] * qkv[(size_t)(b * 32 + kt) * 768 + 512 + h * 32 + c];
        oBf[(size_t)(b * 32 + tq) * 256 + h * 32 + c] = f2bf(o * inv);
    }
}

// ---------------------------------------------------------------------------
// Attention pooling over T: weights = softmax_t(y @ poolW + b); pooled bf16.
// ---------------------------------------------------------------------------
__global__ __launch_bounds__(256) void attn_pool(
    const float* __restrict__ y, const float* __restrict__ poolW,
    const float* __restrict__ poolB, u16* __restrict__ pooledBf)
{
    int b = blockIdx.x, tid = threadIdx.x;
    __shared__ float w[32];
    if (tid < 32) {
        const float* yp = y + (size_t)(b * 32 + tid) * 256;
        float d = 0.0f;
        for (int c = 0; c < 256; c++) d += yp[c] * poolW[c];
        d += poolB[0];
        float mx = d;
        #pragma unroll
        for (int off = 16; off; off >>= 1) mx = fmaxf(mx, __shfl_xor(mx, off, 32));
        float e = __expf(d - mx);
        float s = e;
        #pragma unroll
        for (int off = 16; off; off >>= 1) s += __shfl_xor(s, off, 32);
        w[tid] = e / s;
    }
    __syncthreads();
    int c = tid;
    float acc = 0.0f;
    for (int t = 0; t < 32; t++) acc += w[t] * y[(size_t)(b * 32 + t) * 256 + c];
    pooledBf[(size_t)b * 256 + c] = f2bf(acc);
}

// ---------------------------------------------------------------------------
// Host launcher
// ---------------------------------------------------------------------------
static void launch_cvt(hipStream_t st, const float* src, u16* dst, int n) {
    hipLaunchKernelGGL(cvt_bf16_kernel, dim3((n + 255) / 256), dim3(256), 0, st, src, dst, n);
}
static void launch_gemm(hipStream_t st, const u16* A, const u16* W, const float* bias,
                        const float* rowScale, float* C, u16* Cbf,
                        int R, int N, int K, int flags) {
    int tiles = (R / 32) * (N / 64);
    hipLaunchKernelGGL(gemm_bf16_nt, dim3((tiles + 7) / 8), dim3(256), 0, st,
                       A, W, bias, rowScale, C, Cbf, R, N, K, flags);
}

extern "C" void kernel_launch(void* const* d_in, const int* in_sizes, int n_in,
                              void* d_out, int out_size, void* d_ws, size_t ws_size,
                              hipStream_t stream) {
    const float* drone_feats = (const float*)d_in[0];
    const float* boxes       = (const float*)d_in[1];
    const float* drone_mask  = (const float*)d_in[2];
    const float* in_proj_W   = (const float*)d_in[3];
    const float* in_proj_b   = (const float*)d_in[4];
    const float* gat_lin_W   = (const float*)d_in[5];
    const float* gat_edge_W  = (const float*)d_in[6];
    const float* gat_att_src = (const float*)d_in[7];
    const float* gat_att_dst = (const float*)d_in[8];
    const float* gat_att_edge= (const float*)d_in[9];
    const float* gat_bias    = (const float*)d_in[10];
    const float* gat_ln_s    = (const float*)d_in[11];
    const float* gat_ln_b    = (const float*)d_in[12];
    const float* temp_W      = (const float*)d_in[13];
    const float* temp_b      = (const float*)d_in[14];
    const float* pos_emb     = (const float*)d_in[15];
    const float* qkv_W       = (const float*)d_in[16];
    const float* qkv_b       = (const float*)d_in[17];
    const float* attn_out_W  = (const float*)d_in[18];
    const float* attn_out_b  = (const float*)d_in[19];
    const float* ln1_s       = (const float*)d_in[20];
    const float* ln1_b       = (const float*)d_in[21];
    const float* ln2_s       = (const float*)d_in[22];
    const float* ln2_b       = (const float*)d_in[23];
    const float* ff1_W       = (const float*)d_in[24];
    const float* ff1_b       = (const float*)d_in[25];
    const float* ff2_W       = (const float*)d_in[26];
    const float* ff2_b       = (const float*)d_in[27];
    const float* pool_W      = (const float*)d_in[28];
    const float* pool_b      = (const float*)d_in[29];
    const float* out_W       = (const float*)d_in[30];
    const float* out_b       = (const float*)d_in[31];
    const float* out_ln_s    = (const float*)d_in[32];
    const float* out_ln_b    = (const float*)d_in[33];

    const int RB = 64 * 32 * 64;   // 131072 token rows
    const int BT = 64 * 32;        // 2048 frames

    // bump allocator over d_ws
    char* base = (char*)d_ws;
    size_t off = 0;
    auto alloc = [&](size_t bytes) -> void* {
        void* p = base + off;
        off = (off + bytes + 255) & ~(size_t)255;
        return p;
    };

    u16* inprojWb = (u16*)alloc(256 * 256 * 2);
    u16* gatlinWb = (u16*)alloc(3 * 256 * 256 * 2);   // stored TRANSPOSED per layer
    u16* tempWb   = (u16*)alloc(256 * 256 * 2);
    u16* qkvWb    = (u16*)alloc(2 * 768 * 256 * 2);
    u16* aoutWb   = (u16*)alloc(2 * 256 * 256 * 2);
    u16* ff1Wb    = (u16*)alloc(2 * 512 * 256 * 2);
    u16* ff2Wb    = (u16*)alloc(2 * 256 * 512 * 2);
    u16* outWb    = (u16*)alloc(512 * 256 * 2);

    u16*   featsBf = (u16*)alloc((size_t)RB * 256 * 2);   // reused as alphaT later
    u16*   xBf     = (u16*)alloc((size_t)RB * 256 * 2);
    float* xF      = (float*)alloc((size_t)RB * 256 * 4);
    float* hF      = (float*)alloc((size_t)RB * 256 * 4); // also msg buffer
    u16*   hT      = (u16*)alloc((size_t)RB * 256 * 2);
    float* invsum  = (float*)alloc((size_t)BT * 4 * 64 * 4);
    u16*   frameBf = (u16*)alloc((size_t)BT * 256 * 2);
    float* yF      = (float*)alloc((size_t)BT * 256 * 4);
    u16*   zBf     = (u16*)alloc((size_t)BT * 256 * 2);
    float* qkvF    = (float*)alloc((size_t)BT * 768 * 4);
    u16*   oBf     = (u16*)alloc((size_t)BT * 256 * 2);
    float* ffF     = (float*)alloc((size_t)BT * 512 * 4);
    u16*   ffBf    = (u16*)alloc((size_t)BT * 512 * 2);
    u16*   pooledBf= (u16*)alloc(64 * 256 * 2);
    float* outF    = (float*)alloc(64 * 512 * 4);

    u16* alphaT = featsBf;   // overlay: feats consumed before alphaT is produced

    // --- weight + feature conversion ---
    launch_cvt(stream, in_proj_W,  inprojWb, 256 * 256);
    // gat_lin contracts over the FIRST weight index (x @ W), so store W^T.
    hipLaunchKernelGGL(cvt_bf16_t_kernel, dim3((3 * 256 * 256 + 255) / 256), dim3(256),
                       0, stream, gat_lin_W, gatlinWb, 256, 3 * 256 * 256);
    launch_cvt(stream, temp_W,     tempWb,   256 * 256);
    launch_cvt(stream, qkv_W,      qkvWb,    2 * 768 * 256);
    launch_cvt(stream, attn_out_W, aoutWb,   2 * 256 * 256);
    launch_cvt(stream, ff1_W,      ff1Wb,    2 * 512 * 256);
    launch_cvt(stream, ff2_W,      ff2Wb,    2 * 256 * 512);
    launch_cvt(stream, out_W,      outWb,    512 * 256);
    launch_cvt(stream, drone_feats, featsBf, RB * 256);

    // --- input projection: xF = (feats @ Win^T + b) * mask ---
    launch_gemm(stream, featsBf, inprojWb, in_proj_b, drone_mask, xF, xBf,
                RB, 256, 256, 0);

    // --- GAT layers ---
    for (int l = 0; l < 3; l++) {
        launch_gemm(stream, xBf, gatlinWb + (size_t)l * 256 * 256, nullptr, nullptr,
                    hF, hT, RB, 256, 256, GFLAG_MIR_T64);

        hipLaunchKernelGGL(gat_alpha, dim3(BT), dim3(256), 0, stream,
                           hF, boxes, drone_mask,
                           gat_att_src + l * 256, gat_att_dst + l * 256,
                           gat_att_edge + l * 256, gat_edge_W + (size_t)l * 3 * 256,
                           alphaT, invsum);

        hipLaunchKernelGGL(gat_msg, dim3(BT * 4 / 8), dim3(256), 0, stream,
                           alphaT, hT, invsum, hF);

        hipLaunchKernelGGL(gat_epilogue, dim3(RB / 8), dim3(256), 0, stream,
                           hF, gat_bias + l * 256, gat_ln_s + l * 256, gat_ln_b + l * 256,
                           xF, xBf, RB);
    }

    // --- frame pooling + temporal projection + positional embedding ---
    hipLaunchKernelGGL(frame_pool, dim3(BT), dim3(256), 0, stream, xF, drone_mask, frameBf);
    launch_gemm(stream, frameBf, tempWb, temp_b, nullptr, yF, nullptr, BT, 256, 256, 0);
    hipLaunchKernelGGL(add_pos_kernel, dim3(BT * 256 / 256), dim3(256), 0, stream, yF, pos_emb);

    // --- temporal transformer layers ---
    for (int l = 0; l < 2; l++) {
        hipLaunchKernelGGL(ln_rows, dim3(BT / 8), dim3(256), 0, stream,
                           yF, ln1_s + l * 256, ln1_b + l * 256,
                           (float*)nullptr, zBf, BT, 256, 0);
        launch_gemm(stream, zBf, qkvWb + (size_t)l * 768 * 256, qkv_b + l * 768, nullptr,
                    qkvF, nullptr, BT, 768, 256, 0);
        hipLaunchKernelGGL(temporal_attn, dim3(64), dim3(256), 0, stream, qkvF, oBf);
        launch_gemm(stream, oBf, aoutWb + (size_t)l * 256 * 256, attn_out_b + l * 256, nullptr,
                    yF, nullptr, BT, 256, 256, GFLAG_ADDC);
        hipLaunchKernelGGL(ln_rows, dim3(BT / 8), dim3(256), 0, stream,
                           yF, ln2_s + l * 256, ln2_b + l * 256,
                           (float*)nullptr, zBf, BT, 256, 0);
        launch_gemm(stream, zBf, ff1Wb + (size_t)l * 512 * 256, ff1_b + l * 512, nullptr,
                    ffF, ffBf, BT, 512, 256, GFLAG_RELU);
        launch_gemm(stream, ffBf, ff2Wb + (size_t)l * 256 * 512, ff2_b + l * 256, nullptr,
                    yF, nullptr, BT, 256, 512, GFLAG_ADDC);
    }

    // --- pooling over T, output head, final LN+ReLU into d_out ---
    hipLaunchKernelGGL(attn_pool, dim3(64), dim3(256), 0, stream, yF, pool_W, pool_b, pooledBf);
    launch_gemm(stream, pooledBf, outWb, out_b, nullptr, outF, nullptr, 64, 512, 256, 0);
    hipLaunchKernelGGL(ln_rows, dim3(64 / 8), dim3(256), 0, stream,
                       outF, out_ln_s, out_ln_b, (float*)d_out, (u16*)nullptr, 64, 512, 1);

    (void)in_sizes; (void)n_in; (void)out_size; (void)ws_size;
}
